// DotProductAttention_53815940219143
// MI455X (gfx1250) — matrix operand order repaired and checked by reading, not compiled, not run on hardware
//
#include <hip/hip_runtime.h>
#include <hip/hip_bf16.h>
#include <math.h>

typedef __attribute__((ext_vector_type(16))) _Float16 v16h;
typedef __attribute__((ext_vector_type(8)))  float    v8f;

#define NWIN      4
#define D         64
#define QLEN      1024
#define KVLEN     1024
#define QBLK      128     // q rows per workgroup: 8 waves x 16 rows
#define KVT       64      // kv tile per softmax step (16 WMMAs / iteration)
#define NEGV      (-1000000.0f)

__launch_bounds__(256, 1)
__global__ void attn_flash_wmma(const float* __restrict__ Q,
                                const float* __restrict__ Kmat,
                                const float* __restrict__ Vmat,
                                const int*   __restrict__ vlens,
                                const float* __restrict__ wmask,
                                float* __restrict__ Out)
{
    __shared__ __align__(32) _Float16 ldsK [KVT * D];        // 8 KB  (kv, d) row-major
    __shared__ __align__(32) _Float16 ldsVt[D * KVT];        // 8 KB  (d, kv) transposed
    __shared__ __align__(32) _Float16 ldsP [8 * 16 * KVT];   // 16 KB per-wave P staging

    const int tid  = threadIdx.x;
    const int lane = tid & 31;
    const int wave = tid >> 5;
    const int half = lane >> 4;       // 0: lanes 0-15, 1: lanes 16-31
    const int lidx = lane & 15;

    const int n     = blockIdx.y;
    const int qbase = blockIdx.x * QBLK + wave * 16;

    const int    vlen = vlens[n];
    const int    w    = (n >> 3) & (NWIN - 1);     // n = ((b*4)+w)*8 + h
    const float* wm   = wmask + (size_t)w * QLEN * KVLEN;

    // ---------------- Q tile -> registers in WMMA A-layout (f32 -> f16) ----
    // A 16x32 f16: lane (half) holds row lidx, K elems {lo..lo+15}, lo=half*16
    v16h aQ[2];
    {
        const float* qrow = Q + ((size_t)n * QLEN + qbase + lidx) * D;
        #pragma unroll
        for (int t = 0; t < 2; ++t) {
            const int   base = t * 32 + half * 16;
            const float4 a0 = *(const float4*)(qrow + base);
            const float4 a1 = *(const float4*)(qrow + base + 4);
            const float4 a2 = *(const float4*)(qrow + base + 8);
            const float4 a3 = *(const float4*)(qrow + base + 12);
            aQ[t][0]  = (_Float16)a0.x; aQ[t][1]  = (_Float16)a0.y;
            aQ[t][2]  = (_Float16)a0.z; aQ[t][3]  = (_Float16)a0.w;
            aQ[t][4]  = (_Float16)a1.x; aQ[t][5]  = (_Float16)a1.y;
            aQ[t][6]  = (_Float16)a1.z; aQ[t][7]  = (_Float16)a1.w;
            aQ[t][8]  = (_Float16)a2.x; aQ[t][9]  = (_Float16)a2.y;
            aQ[t][10] = (_Float16)a2.z; aQ[t][11] = (_Float16)a2.w;
            aQ[t][12] = (_Float16)a3.x; aQ[t][13] = (_Float16)a3.y;
            aQ[t][14] = (_Float16)a3.z; aQ[t][15] = (_Float16)a3.w;
        }
    }

    // ---------------- accumulators ----------------
    v8f   Oacc[4];
    float m[8], l[8];
    #pragma unroll
    for (int c = 0; c < 4; ++c)
        #pragma unroll
        for (int r = 0; r < 8; ++r) Oacc[c][r] = 0.0f;
    #pragma unroll
    for (int r = 0; r < 8; ++r) { m[r] = -3.0e38f; l[r] = 0.0f; }

    // ---------------- software-pipelined K/V staging -----------------------
    // 64x64 f32 tile, 256 threads -> 16 f32 per thread per tensor (4x float4)
    const float* Kbase = Kmat + (size_t)n * KVLEN * D;
    const float* Vbase = Vmat + (size_t)n * KVLEN * D;
    const int    srow  = tid >> 2;          // 0..63
    const int    scol  = (tid & 3) * 16;    // 0,16,32,48

    float4 kreg[4], vreg[4];
    {   // prologue fetch of tile 0
        const float* kp = Kbase + (size_t)srow * D + scol;
        const float* vp = Vbase + (size_t)srow * D + scol;
        #pragma unroll
        for (int i = 0; i < 4; ++i) {
            kreg[i] = *(const float4*)(kp + i * 4);
            vreg[i] = *(const float4*)(vp + i * 4);
        }
    }

    for (int kv0 = 0; kv0 < KVLEN; kv0 += KVT) {
        // ---- commit pipelined registers to LDS (f32 -> f16) --------------
        __syncthreads();                       // previous tile's readers done
        {
            v16h krow;
            #pragma unroll
            for (int i = 0; i < 4; ++i) {
                krow[i * 4 + 0] = (_Float16)kreg[i].x;
                krow[i * 4 + 1] = (_Float16)kreg[i].y;
                krow[i * 4 + 2] = (_Float16)kreg[i].z;
                krow[i * 4 + 3] = (_Float16)kreg[i].w;
            }
            *(v16h*)&ldsK[srow * D + scol] = krow;      // one 32B LDS store

            #pragma unroll
            for (int i = 0; i < 4; ++i) {               // transposed scatter
                ldsVt[(scol + i * 4 + 0) * KVT + srow] = (_Float16)vreg[i].x;
                ldsVt[(scol + i * 4 + 1) * KVT + srow] = (_Float16)vreg[i].y;
                ldsVt[(scol + i * 4 + 2) * KVT + srow] = (_Float16)vreg[i].z;
                ldsVt[(scol + i * 4 + 3) * KVT + srow] = (_Float16)vreg[i].w;
            }
        }
        __syncthreads();

        // ---- issue next tile's global loads; latency hides under compute --
        if (kv0 + KVT < KVLEN) {
            const float* kp = Kbase + (size_t)(kv0 + KVT + srow) * D + scol;
            const float* vp = Vbase + (size_t)(kv0 + KVT + srow) * D + scol;
            #pragma unroll
            for (int i = 0; i < 4; ++i) {
                kreg[i] = *(const float4*)(kp + i * 4);
                vreg[i] = *(const float4*)(vp + i * 4);
            }
        }

        // ---- scores S = Q K^T : 4 n-tiles, K-depth 64 ---------------------
        v8f S[4];
        #pragma unroll
        for (int t = 0; t < 4; ++t) {
            v8f acc = {};
            #pragma unroll
            for (int c = 0; c < 2; ++c) {
                // B 32x16: lane holds kv col (t*16+lidx); d-dim contiguous
                const v16h b = *(const v16h*)&ldsK[(t * 16 + lidx) * D + c * 32 + half * 16];
                acc = __builtin_amdgcn_wmma_f32_16x16x32_f16(
                        false, aQ[c], false, b, (short)0, acc, false, false);
            }
            S[t] = acc;
        }

        // ---- scale + window mask + valid_len mask -------------------------
        #pragma unroll
        for (int r = 0; r < 8; ++r) {
            const int    q  = qbase + r + half * 8;
            const float* wp = wm + (size_t)q * KVLEN + kv0 + lidx;
            #pragma unroll
            for (int t = 0; t < 4; ++t) {
                const int   kg = kv0 + t * 16 + lidx;
                const float s  = S[t][r] * 0.125f + wp[t * 16];
                S[t][r] = (kg < vlen) ? s : NEGV;
            }
        }

        // ---- online softmax (reductions stay inside each 16-lane half) ----
        float alpha[8];
        #pragma unroll
        for (int r = 0; r < 8; ++r) {
            float mx = fmaxf(fmaxf(S[0][r], S[1][r]), fmaxf(S[2][r], S[3][r]));
            #pragma unroll
            for (int off = 1; off < 16; off <<= 1)
                mx = fmaxf(mx, __shfl_xor(mx, off, 32));
            const float mnew = fmaxf(m[r], mx);
            float rs = 0.0f;
            #pragma unroll
            for (int t = 0; t < 4; ++t) {
                const float p = __expf(S[t][r] - mnew);
                S[t][r] = p;
                rs += p;
            }
            #pragma unroll
            for (int off = 1; off < 16; off <<= 1)
                rs += __shfl_xor(rs, off, 32);
            alpha[r] = __expf(m[r] - mnew);
            l[r]     = l[r] * alpha[r] + rs;
            m[r]     = mnew;
        }
        #pragma unroll
        for (int c = 0; c < 4; ++c)
            #pragma unroll
            for (int r = 0; r < 8; ++r) Oacc[c][r] *= alpha[r];

        // ---- P: C-layout -> LDS -> A-layout (per-wave slab) ---------------
        _Float16* pP = &ldsP[wave * 16 * KVT];
        #pragma unroll
        for (int t = 0; t < 4; ++t)
            #pragma unroll
            for (int r = 0; r < 8; ++r)
                pP[(r + half * 8) * KVT + t * 16 + lidx] = (_Float16)S[t][r];
        asm volatile("s_wait_dscnt 0" ::: "memory");    // same-wave LDS RAW

        v16h aP[2];
        {
            const _Float16* prow = &pP[lidx * KVT];
            #pragma unroll
            for (int kc = 0; kc < 2; ++kc) {
                const int lo = kc * 32 + half * 16;
                #pragma unroll
                for (int i = 0; i < 8; ++i) aP[kc][i]     = prow[lo + i];
                #pragma unroll
                for (int i = 0; i < 8; ++i) aP[kc][8 + i] = prow[lo + 8 + i];
            }
        }

        // ---- O += P V : 4 d-chunks x K-depth 64 ---------------------------
        #pragma unroll
        for (int c = 0; c < 4; ++c) {
            #pragma unroll
            for (int kc = 0; kc < 2; ++kc) {
                // B 32x16 from V^T: lane holds d col (c*16+lidx); kv contiguous
                const v16h b = *(const v16h*)&ldsVt[(c * 16 + lidx) * KVT + kc * 32 + half * 16];
                Oacc[c] = __builtin_amdgcn_wmma_f32_16x16x32_f16(
                            false, aP[kc], false, b, (short)0, Oacc[c], false, false);
            }
        }
    }

    // ---------------- normalize and write out ------------------------------
    float* orow = Out + ((size_t)n * QLEN + qbase) * D;
    #pragma unroll
    for (int r = 0; r < 8; ++r) {
        const float inv = 1.0f / l[r];
        const int   q   = r + half * 8;
        #pragma unroll
        for (int c = 0; c < 4; ++c)
            orow[(size_t)q * D + c * 16 + lidx] = Oacc[c][r] * inv;
    }
}

extern "C" void kernel_launch(void* const* d_in, const int* in_sizes, int n_in,
                              void* d_out, int out_size, void* d_ws, size_t ws_size,
                              hipStream_t stream)
{
    const float* Q     = (const float*)d_in[0];
    const float* K     = (const float*)d_in[1];
    const float* V     = (const float*)d_in[2];
    const int*   vlens = (const int*)  d_in[3];
    const float* wmask = (const float*)d_in[4];
    float*       Out   = (float*)d_out;

    const int n = in_sizes[3];                 // 256 heads
    dim3 grid(QLEN / QBLK, n);
    attn_flash_wmma<<<grid, 256, 0, stream>>>(Q, K, V, vlens, wmask, Out);
}
